// GOGraphEncoderUnified_43868795961639
// MI455X (gfx1250) — matrix operand order repaired and checked by reading, not compiled, not run on hardware
//
#include <hip/hip_runtime.h>
#include <math.h>

typedef __bf16 bhalf;
typedef __attribute__((ext_vector_type(8)))  __bf16 v8bf;
typedef __attribute__((ext_vector_type(16))) __bf16 v16bf;
typedef __attribute__((ext_vector_type(8)))  float  v8f;

#define NN   45000
#define EE   150000
#define ET   (EE + NN)
#define DD   2560
#define HIDN 512
#define NH   8
#define NC   64
#define NQQ  200
#define HDD  320
#define KPAD 45056   // padded key stride for vt

// ---------------------------------------------------------------------------
// helpers
// ---------------------------------------------------------------------------
__device__ __forceinline__ float gelu_f(float x) {
  return 0.5f * x * (1.0f + erff(x * 0.70710678118654752f));
}

// order-preserving encoding for float atomic-max on unsigned
__device__ __forceinline__ unsigned fenc(float f) {
  unsigned u = __float_as_uint(f);
  return (u & 0x80000000u) ? ~u : (u | 0x80000000u);
}
__device__ __forceinline__ float fdec(unsigned e) {
  unsigned u = (e & 0x80000000u) ? (e & 0x7FFFFFFFu) : ~e;
  return __uint_as_float(u);
}

// A fragment: 16x32 bf16, row-major source, lane l -> row (l&15), half = l>>4
// per ISA: lanes 0-15 hold K = {half*8 .. +8} and {16+half*8 .. +8}
__device__ __forceinline__ v16bf load_afrag(const bhalf* rowk, int half) {
  v8bf lo = *(const v8bf*)(rowk + half * 8);
  v8bf hi = *(const v8bf*)(rowk + 16 + half * 8);
  v16bf f;
#pragma unroll
  for (int i = 0; i < 8; ++i) { f[i] = lo[i]; f[8 + i] = hi[i]; }
  return f;
}

// B fragment: 32x16 bf16 from Bt[N,K] row-major; lane l -> col (l&15),
// holds K = {half*8 .. +8} and {16+half*8 .. +8}
__device__ __forceinline__ v16bf load_bfrag(const bhalf* rowk, int half) {
  v8bf lo = *(const v8bf*)(rowk + half * 8);
  v8bf hi = *(const v8bf*)(rowk + 16 + half * 8);
  v16bf f;
#pragma unroll
  for (int i = 0; i < 8; ++i) { f[i] = lo[i]; f[8 + i] = hi[i]; }
  return f;
}

// ---------------------------------------------------------------------------
// conversion / utility kernels
// ---------------------------------------------------------------------------
__global__ void f32_to_bf16_kernel(const float* __restrict__ in, bhalf* __restrict__ out, long long n) {
  long long i = (long long)blockIdx.x * blockDim.x + threadIdx.x;
  if (i < n) out[i] = (bhalf)in[i];
}

// in [K,N] f32 -> out [N,K] bf16
__global__ void transpose_to_bf16_kernel(const float* __restrict__ in, bhalf* __restrict__ out, int K, int N) {
  long long i = (long long)blockIdx.x * blockDim.x + threadIdx.x;
  if (i >= (long long)K * N) return;
  int k = (int)(i / N), n = (int)(i % N);
  out[(size_t)n * K + k] = (bhalf)in[i];
}

__global__ void fill_u32_kernel(unsigned* __restrict__ p, unsigned v, int n) {
  int i = blockIdx.x * blockDim.x + threadIdx.x;
  if (i < n) p[i] = v;
}

// ---------------------------------------------------------------------------
// generic bf16 WMMA GEMM: out = epi(A[M,K] @ Bt[N,K]^T + bias)
// EPI: 0=none, 1=bias, 2=bias+gelu
// OUTM: 0=f32 row-major, 1=bf16 row-major, 2=bf16 transposed ([N, ldo])
// block = 256 thr (8 waves = 4x2); wave tile 32x32 (2x2 WMMA); block tile 128x64
// load:WMMA ratio 1.0 (2 A-frags + 2 B-frags -> 4 WMMAs per K-step)
// ---------------------------------------------------------------------------
template <int EPI, int OUTM>
__global__ __launch_bounds__(256) void gemm_wmma_kernel(
    const bhalf* __restrict__ A, const bhalf* __restrict__ Bt,
    const float* __restrict__ bias, void* __restrict__ outp,
    int M, int N, int K, int lda, int ldb, int ldo) {
  const int lane = threadIdx.x & 31;
  const int w    = threadIdx.x >> 5;
  const int half = lane >> 4;
  const int l15  = lane & 15;
  const int mBase = blockIdx.y * 128 + (w >> 1) * 32;
  const int nBase = blockIdx.x * 64 + (w & 1) * 32;

  const int arow0 = min(mBase + l15, M - 1);
  const int arow1 = min(mBase + 16 + l15, M - 1);
  const bhalf* aptr0 = A + (size_t)arow0 * lda;
  const bhalf* aptr1 = A + (size_t)arow1 * lda;
  const bhalf* bptr0 = Bt + (size_t)(nBase + l15) * ldb;
  const bhalf* bptr1 = Bt + (size_t)(nBase + 16 + l15) * ldb;

  v8f acc00 = {}; v8f acc01 = {};
  v8f acc10 = {}; v8f acc11 = {};
  for (int kb = 0; kb < K; kb += 32) {
    if (kb + 32 < K) {
      __builtin_prefetch(aptr0 + kb + 32, 0, 3);
      __builtin_prefetch(aptr1 + kb + 32, 0, 3);
    }
    v16bf a0 = load_afrag(aptr0 + kb, half);
    v16bf a1 = load_afrag(aptr1 + kb, half);
    v16bf b0 = load_bfrag(bptr0 + kb, half);
    v16bf b1 = load_bfrag(bptr1 + kb, half);
    acc00 = __builtin_amdgcn_wmma_f32_16x16x32_bf16(false, a0, false, b0, (short)0, acc00, false, false);
    acc01 = __builtin_amdgcn_wmma_f32_16x16x32_bf16(false, a0, false, b1, (short)0, acc01, false, false);
    acc10 = __builtin_amdgcn_wmma_f32_16x16x32_bf16(false, a1, false, b0, (short)0, acc10, false, false);
    acc11 = __builtin_amdgcn_wmma_f32_16x16x32_bf16(false, a1, false, b1, (short)0, acc11, false, false);
  }

  const int c0 = nBase + l15, c1 = c0 + 16;
  const float bv0 = (EPI != 0) ? bias[c0] : 0.f;
  const float bv1 = (EPI != 0) ? bias[c1] : 0.f;
#pragma unroll
  for (int r = 0; r < 8; ++r) {
    const int row0 = mBase + r + half * 8;
    const int row1 = row0 + 16;
    float v00 = acc00[r] + bv0, v01 = acc01[r] + bv1;
    float v10 = acc10[r] + bv0, v11 = acc11[r] + bv1;
    if (EPI == 2) { v00 = gelu_f(v00); v01 = gelu_f(v01); v10 = gelu_f(v10); v11 = gelu_f(v11); }
    if (OUTM == 0) {
      float* o = (float*)outp;
      if (row0 < M) { o[(size_t)row0 * ldo + c0] = v00; o[(size_t)row0 * ldo + c1] = v01; }
      if (row1 < M) { o[(size_t)row1 * ldo + c0] = v10; o[(size_t)row1 * ldo + c1] = v11; }
    } else if (OUTM == 1) {
      bhalf* o = (bhalf*)outp;
      if (row0 < M) { o[(size_t)row0 * ldo + c0] = (bhalf)v00; o[(size_t)row0 * ldo + c1] = (bhalf)v01; }
      if (row1 < M) { o[(size_t)row1 * ldo + c0] = (bhalf)v10; o[(size_t)row1 * ldo + c1] = (bhalf)v11; }
    } else {
      bhalf* o = (bhalf*)outp;
      if (row0 < M) { o[(size_t)c0 * ldo + row0] = (bhalf)v00; o[(size_t)c1 * ldo + row0] = (bhalf)v01; }
      if (row1 < M) { o[(size_t)c0 * ldo + row1] = (bhalf)v10; o[(size_t)c1 * ldo + row1] = (bhalf)v11; }
    }
  }
}

// ---------------------------------------------------------------------------
// GAT kernels
// ---------------------------------------------------------------------------
__global__ void gat_logits_kernel(const float* __restrict__ h, const float* __restrict__ as,
                                  const float* __restrict__ ad, float* __restrict__ ssrc,
                                  float* __restrict__ sdst) {
  int i = blockIdx.x * blockDim.x + threadIdx.x;
  if (i >= NN * NH) return;
  int node = i >> 3, hh = i & 7;
  const float* hp = h + (size_t)node * HIDN + hh * NC;
  float s1 = 0.f, s2 = 0.f;
#pragma unroll 8
  for (int c = 0; c < NC; ++c) {
    float v = hp[c];
    s1 += v * as[hh * NC + c];
    s2 += v * ad[hh * NC + c];
  }
  ssrc[i] = s1; sdst[i] = s2;
}

__global__ void gat_edge_max_kernel(const int* __restrict__ ei, const float* __restrict__ ssrc,
                                    const float* __restrict__ sdst, unsigned* __restrict__ emax) {
  int i = blockIdx.x * blockDim.x + threadIdx.x;
  if (i >= ET * NH) return;
  int e = i >> 3, hh = i & 7;
  int s = (e < EE) ? ei[e] : (e - EE);
  int d = (e < EE) ? ei[EE + e] : (e - EE);
  float x = ssrc[s * NH + hh] + sdst[d * NH + hh];
  x = (x > 0.f) ? x : 0.2f * x;               // leaky_relu 0.2
  atomicMax(&emax[d * NH + hh], fenc(x));
}

__global__ void gat_edge_exp_kernel(const int* __restrict__ ei, const float* __restrict__ ssrc,
                                    const float* __restrict__ sdst, const unsigned* __restrict__ emax,
                                    float* __restrict__ exbuf, float* __restrict__ denom) {
  int i = blockIdx.x * blockDim.x + threadIdx.x;
  if (i >= ET * NH) return;
  int e = i >> 3, hh = i & 7;
  int s = (e < EE) ? ei[e] : (e - EE);
  int d = (e < EE) ? ei[EE + e] : (e - EE);
  float x = ssrc[s * NH + hh] + sdst[d * NH + hh];
  x = (x > 0.f) ? x : 0.2f * x;
  float m = fdec(emax[d * NH + hh]);
  if (!(m > -3.0e38f && m < 3.0e38f)) m = 0.f;  // mirror where(isfinite, max, 0)
  float ex = __expf(x - m);
  exbuf[i] = ex;
  atomicAdd(&denom[d * NH + hh], ex);
}

__global__ void gat_edge_agg_kernel(const int* __restrict__ ei, const float* __restrict__ hraw,
                                    const float* __restrict__ exbuf, const float* __restrict__ denom,
                                    float* __restrict__ agg) {
  int i = blockIdx.x * blockDim.x + threadIdx.x;
  if (i >= ET * NH) return;
  int e = i >> 3, hh = i & 7;
  int s = (e < EE) ? ei[e] : (e - EE);
  int d = (e < EE) ? ei[EE + e] : (e - EE);
  float alpha = exbuf[i] / (denom[d * NH + hh] + 1e-16f);
  const float* hp = hraw + (size_t)s * HIDN + hh * NC;
  float* op = agg + (size_t)d * HIDN + hh * NC;
#pragma unroll 8
  for (int c = 0; c < NC; ++c) atomicAdd(&op[c], hp[c] * alpha);
}

__global__ void bias_elu_bf16_kernel(const float* __restrict__ agg, const float* __restrict__ b,
                                     bhalf* __restrict__ out) {
  long long i = (long long)blockIdx.x * blockDim.x + threadIdx.x;
  if (i >= (long long)NN * HIDN) return;
  int col = (int)(i & (HIDN - 1));
  float v = agg[i] + b[col];
  v = (v > 0.f) ? v : (__expf(v) - 1.f);      // elu
  out[i] = (bhalf)v;
}

// ---------------------------------------------------------------------------
// flash cross-attention: 16 queries x 1 head per block, 10 waves, each wave
// owns two 16-wide HD tiles (HD=320). Streams keys in chunks of 32.
// Qb [NQQ,DD] bf16, Kb [NN,DD] bf16, Vt [DD, KPAD] bf16 (head-major rows).
// Ctx out bf16 [NQQ, DD].
// ---------------------------------------------------------------------------
__global__ __launch_bounds__(320) void flash_attn_kernel(const bhalf* __restrict__ Qb,
                                                         const bhalf* __restrict__ Kb,
                                                         const bhalf* __restrict__ Vt,
                                                         bhalf* __restrict__ Ctx) {
  const int h     = blockIdx.x;        // head
  const int qBase = blockIdx.y * 16;   // query tile
  const int w     = threadIdx.x >> 5;  // wave 0..9
  const int lane  = threadIdx.x & 31;
  const int half  = lane >> 4;
  const int l15   = lane & 15;
  const float scale = 0.05590169943749474f;  // 1/sqrt(320)

  __shared__ float sP[10][16 * 32];    // per-wave private P staging
  float* myP = &sP[w][0];

  // Q fragments for all 10 K-chunks of HD
  const int qrow = min(qBase + l15, NQQ - 1);
  const bhalf* qp = Qb + (size_t)qrow * DD + h * HDD;
  v16bf qf[10];
#pragma unroll
  for (int c = 0; c < 10; ++c) qf[c] = load_afrag(qp + c * 32, half);

  const int d0 = w * 32;               // this wave's HD tiles: d0, d0+16
  const bhalf* vp0 = Vt + (size_t)(h * HDD + d0 + l15) * KPAD;
  const bhalf* vp1 = Vt + (size_t)(h * HDD + d0 + 16 + l15) * KPAD;

  v8f o0 = {}; v8f o1 = {};
  float mrun[8], lrun[8];
#pragma unroll
  for (int r = 0; r < 8; ++r) { mrun[r] = -1e30f; lrun[r] = 0.f; }

  for (int kb = 0; kb < NN; kb += 32) {
    // ---- S = Q K^T for two 16-key tiles
    const int kn0 = min(kb + l15, NN - 1);
    const int kn1 = min(kb + 16 + l15, NN - 1);
    const bhalf* kp0 = Kb + (size_t)kn0 * DD + h * HDD;
    const bhalf* kp1 = Kb + (size_t)kn1 * DD + h * HDD;
    v8f s0 = {}; v8f s1 = {};
#pragma unroll
    for (int c = 0; c < 10; ++c) {
      v16bf b0 = load_bfrag(kp0 + c * 32, half);
      v16bf b1 = load_bfrag(kp1 + c * 32, half);
      s0 = __builtin_amdgcn_wmma_f32_16x16x32_bf16(false, qf[c], false, b0, (short)0, s0, false, false);
      s1 = __builtin_amdgcn_wmma_f32_16x16x32_bf16(false, qf[c], false, b1, (short)0, s1, false, false);
    }
    // ---- scale + mask invalid keys
    const bool vok0 = (kb + l15) < NN;
    const bool vok1 = (kb + 16 + l15) < NN;
#pragma unroll
    for (int r = 0; r < 8; ++r) {
      s0[r] = vok0 ? s0[r] * scale : -1e30f;
      s1[r] = vok1 ? s1[r] * scale : -1e30f;
    }
    // ---- running softmax (rows r+8*half, 16-lane butterfly within half)
    float rmax[8];
#pragma unroll
    for (int r = 0; r < 8; ++r) rmax[r] = fmaxf(s0[r], s1[r]);
#pragma unroll
    for (int off = 1; off < 16; off <<= 1)
#pragma unroll
      for (int r = 0; r < 8; ++r) rmax[r] = fmaxf(rmax[r], __shfl_xor(rmax[r], off, 32));
    float corr[8];
#pragma unroll
    for (int r = 0; r < 8; ++r) {
      float mn = fmaxf(mrun[r], rmax[r]);
      corr[r] = __expf(mrun[r] - mn);
      mrun[r] = mn;
    }
    float rsum[8];
#pragma unroll
    for (int r = 0; r < 8; ++r) {
      s0[r] = __expf(s0[r] - mrun[r]);
      s1[r] = __expf(s1[r] - mrun[r]);
      rsum[r] = s0[r] + s1[r];
    }
#pragma unroll
    for (int off = 1; off < 16; off <<= 1)
#pragma unroll
      for (int r = 0; r < 8; ++r) rsum[r] += __shfl_xor(rsum[r], off, 32);
#pragma unroll
    for (int r = 0; r < 8; ++r) {
      lrun[r] = lrun[r] * corr[r] + rsum[r];
      o0[r] *= corr[r];
      o1[r] *= corr[r];
    }
    // ---- P (D-layout) -> LDS -> A-fragment layout
#pragma unroll
    for (int r = 0; r < 8; ++r) {
      myP[(r + 8 * half) * 32 + l15]      = s0[r];
      myP[(r + 8 * half) * 32 + 16 + l15] = s1[r];
    }
    v16bf pf;
#pragma unroll
    for (int j = 0; j < 8; ++j) {
      int k0 = (j < 4) ? (half * 8 + 2 * j) : (16 + half * 8 + 2 * (j - 4));
      pf[2 * j]     = (bhalf)myP[l15 * 32 + k0];
      pf[2 * j + 1] = (bhalf)myP[l15 * 32 + k0 + 1];
    }
    // ---- O += P V
    v16bf vf0 = load_bfrag(vp0 + kb, half);
    v16bf vf1 = load_bfrag(vp1 + kb, half);
    o0 = __builtin_amdgcn_wmma_f32_16x16x32_bf16(false, pf, false, vf0, (short)0, o0, false, false);
    o1 = __builtin_amdgcn_wmma_f32_16x16x32_bf16(false, pf, false, vf1, (short)0, o1, false, false);
  }

  // ---- normalize and store
#pragma unroll
  for (int r = 0; r < 8; ++r) {
    int row = qBase + r + 8 * half;
    if (row >= NQQ) continue;
    float inv = 1.f / lrun[r];
    Ctx[(size_t)row * DD + h * HDD + d0 + l15]      = (bhalf)(o0[r] * inv);
    Ctx[(size_t)row * DD + h * HDD + d0 + 16 + l15] = (bhalf)(o1[r] * inv);
  }
}

// ---------------------------------------------------------------------------
// out = LayerNorm(x + y) * g + b ; optional bf16 copy
// ---------------------------------------------------------------------------
__global__ __launch_bounds__(256) void add_ln_kernel(const float* __restrict__ x,
                                                     const float* __restrict__ y,
                                                     const float* __restrict__ g,
                                                     const float* __restrict__ b,
                                                     float* __restrict__ outf,
                                                     bhalf* __restrict__ outb) {
  const int row = blockIdx.x;
  const float* xp = x + (size_t)row * DD;
  const float* yp = y + (size_t)row * DD;
  __shared__ float red[256];
  float s = 0.f;
  for (int i = threadIdx.x; i < DD; i += 256) s += xp[i] + yp[i];
  red[threadIdx.x] = s; __syncthreads();
  for (int o = 128; o > 0; o >>= 1) { if (threadIdx.x < o) red[threadIdx.x] += red[threadIdx.x + o]; __syncthreads(); }
  const float mean = red[0] * (1.0f / DD);
  __syncthreads();
  float v = 0.f;
  for (int i = threadIdx.x; i < DD; i += 256) { float d = xp[i] + yp[i] - mean; v += d * d; }
  red[threadIdx.x] = v; __syncthreads();
  for (int o = 128; o > 0; o >>= 1) { if (threadIdx.x < o) red[threadIdx.x] += red[threadIdx.x + o]; __syncthreads(); }
  const float rstd = rsqrtf(red[0] * (1.0f / DD) + 1e-5f);
  for (int i = threadIdx.x; i < DD; i += 256) {
    float o = (xp[i] + yp[i] - mean) * rstd * g[i] + b[i];
    outf[(size_t)row * DD + i] = o;
    if (outb) outb[(size_t)row * DD + i] = (bhalf)o;
  }
}

// ---------------------------------------------------------------------------
// host orchestration
// ---------------------------------------------------------------------------
static inline unsigned cdiv(long long a, long long b) { return (unsigned)((a + b - 1) / b); }

extern "C" void kernel_launch(void* const* d_in, const int* in_sizes, int n_in,
                              void* d_out, int out_size, void* d_ws, size_t ws_size,
                              hipStream_t stream) {
  (void)in_sizes; (void)n_in; (void)out_size; (void)ws_size;
  const float* x   = (const float*)d_in[0];
  const int*   ei  = (const int*)d_in[1];
  const float* gw[3]  = {(const float*)d_in[2],  (const float*)d_in[6],  (const float*)d_in[10]};
  const float* gas[3] = {(const float*)d_in[3],  (const float*)d_in[7],  (const float*)d_in[11]};
  const float* gad[3] = {(const float*)d_in[4],  (const float*)d_in[8],  (const float*)d_in[12]};
  const float* gb[3]  = {(const float*)d_in[5],  (const float*)d_in[9],  (const float*)d_in[13]};
  const float* pw1 = (const float*)d_in[14]; const float* pb1 = (const float*)d_in[15];
  const float* pw2 = (const float*)d_in[16]; const float* pb2 = (const float*)d_in[17];
  const float* queries = (const float*)d_in[18];
  const float* wq = (const float*)d_in[19]; const float* wk = (const float*)d_in[20];
  const float* wv = (const float*)d_in[21]; const float* wo = (const float*)d_in[22];
  const float* bq = (const float*)d_in[23]; const float* bk = (const float*)d_in[24];
  const float* bv = (const float*)d_in[25]; const float* bo = (const float*)d_in[26];
  const float* ln1g = (const float*)d_in[27]; const float* ln2g = (const float*)d_in[28];
  const float* ln1b = (const float*)d_in[29]; const float* ln2b = (const float*)d_in[30];
  const float* fw1 = (const float*)d_in[31]; const float* fb1 = (const float*)d_in[32];
  const float* fw2 = (const float*)d_in[33]; const float* fb2 = (const float*)d_in[34];
  float* out = (float*)d_out;

  char* ws = (char*)d_ws;
  size_t off = 0;
  auto alloc = [&](size_t bytes) -> char* {
    char* p = ws + off;
    off += (bytes + 255) & ~(size_t)255;
    return p;
  };

  bhalf* xb    = (bhalf*)alloc((size_t)NN * DD * 2);
  bhalf* gw0t  = (bhalf*)alloc((size_t)HIDN * DD * 2);
  bhalf* gw1t  = (bhalf*)alloc((size_t)HIDN * HIDN * 2);
  bhalf* gw2t  = (bhalf*)alloc((size_t)HIDN * HIDN * 2);
  bhalf* pw1t  = (bhalf*)alloc((size_t)HIDN * HIDN * 2);
  bhalf* pw2t  = (bhalf*)alloc((size_t)DD * HIDN * 2);
  bhalf* wqt   = (bhalf*)alloc((size_t)DD * DD * 2);
  bhalf* wkt   = (bhalf*)alloc((size_t)DD * DD * 2);
  bhalf* wvt   = (bhalf*)alloc((size_t)DD * DD * 2);
  bhalf* wot   = (bhalf*)alloc((size_t)DD * DD * 2);
  bhalf* fw1t  = (bhalf*)alloc((size_t)4 * DD * DD * 2);
  bhalf* fw2t  = (bhalf*)alloc((size_t)DD * 4 * DD * 2);
  float* hraw  = (float*)alloc((size_t)NN * HIDN * 4);
  float* agg   = (float*)alloc((size_t)NN * HIDN * 4);
  bhalf* hact  = (bhalf*)alloc((size_t)NN * HIDN * 2);
  float* ssrc  = (float*)alloc((size_t)NN * NH * 4);
  float* sdst  = (float*)alloc((size_t)NN * NH * 4);
  unsigned* emax = (unsigned*)alloc((size_t)NN * NH * 4);
  float* denom = (float*)alloc((size_t)NN * NH * 4);
  float* exbuf = (float*)alloc((size_t)ET * NH * 4);
  bhalf* t1b   = (bhalf*)alloc((size_t)NN * HIDN * 2);
  bhalf* gob   = (bhalf*)alloc((size_t)NN * DD * 2);
  bhalf* qinb  = (bhalf*)alloc((size_t)NQQ * DD * 2);
  bhalf* qb    = (bhalf*)alloc((size_t)NQQ * DD * 2);
  bhalf* kbf   = (bhalf*)alloc((size_t)NN * DD * 2);
  bhalf* vt    = (bhalf*)alloc((size_t)DD * KPAD * 2);
  bhalf* ctxb  = (bhalf*)alloc((size_t)NQQ * DD * 2);
  float* attnout = (float*)alloc((size_t)NQQ * DD * 4);
  float* a1    = (float*)alloc((size_t)NQQ * DD * 4);
  bhalf* a1b   = (bhalf*)alloc((size_t)NQQ * DD * 2);
  bhalf* f1b   = (bhalf*)alloc((size_t)NQQ * 4 * DD * 2);
  float* ffn2  = (float*)alloc((size_t)NQQ * DD * 4);

  const dim3 T(256);

  // ---- conversions / weight transposes (re-done every call: deterministic)
  f32_to_bf16_kernel<<<cdiv((long long)NN * DD, 256), T, 0, stream>>>(x, xb, (long long)NN * DD);
  f32_to_bf16_kernel<<<cdiv((long long)NQQ * DD, 256), T, 0, stream>>>(queries, qinb, (long long)NQQ * DD);
  transpose_to_bf16_kernel<<<cdiv((long long)DD * HIDN, 256), T, 0, stream>>>(gw[0], gw0t, DD, HIDN);
  transpose_to_bf16_kernel<<<cdiv((long long)HIDN * HIDN, 256), T, 0, stream>>>(gw[1], gw1t, HIDN, HIDN);
  transpose_to_bf16_kernel<<<cdiv((long long)HIDN * HIDN, 256), T, 0, stream>>>(gw[2], gw2t, HIDN, HIDN);
  transpose_to_bf16_kernel<<<cdiv((long long)HIDN * HIDN, 256), T, 0, stream>>>(pw1, pw1t, HIDN, HIDN);
  transpose_to_bf16_kernel<<<cdiv((long long)HIDN * DD, 256), T, 0, stream>>>(pw2, pw2t, HIDN, DD);
  transpose_to_bf16_kernel<<<cdiv((long long)DD * DD, 256), T, 0, stream>>>(wq, wqt, DD, DD);
  transpose_to_bf16_kernel<<<cdiv((long long)DD * DD, 256), T, 0, stream>>>(wk, wkt, DD, DD);
  transpose_to_bf16_kernel<<<cdiv((long long)DD * DD, 256), T, 0, stream>>>(wv, wvt, DD, DD);
  transpose_to_bf16_kernel<<<cdiv((long long)DD * DD, 256), T, 0, stream>>>(wo, wot, DD, DD);
  transpose_to_bf16_kernel<<<cdiv((long long)DD * 4 * DD, 256), T, 0, stream>>>(fw1, fw1t, DD, 4 * DD);
  transpose_to_bf16_kernel<<<cdiv((long long)4 * DD * DD, 256), T, 0, stream>>>(fw2, fw2t, 4 * DD, DD);

  // ---- 3 GAT layers
  for (int l = 0; l < 3; ++l) {
    const bhalf* Ain  = (l == 0) ? xb : hact;
    const bhalf* Wt   = (l == 0) ? gw0t : (l == 1) ? gw1t : gw2t;
    const int    Kdim = (l == 0) ? DD : HIDN;
    dim3 G(HIDN / 64, cdiv(NN, 128));
    gemm_wmma_kernel<0, 0><<<G, T, 0, stream>>>(Ain, Wt, nullptr, hraw, NN, HIDN, Kdim, Kdim, Kdim, HIDN);
    gat_logits_kernel<<<cdiv((long long)NN * NH, 256), T, 0, stream>>>(hraw, gas[l], gad[l], ssrc, sdst);
    fill_u32_kernel<<<cdiv((long long)NN * NH, 256), T, 0, stream>>>(emax, 0x007FFFFFu, NN * NH); // enc(-inf)
    hipMemsetAsync(denom, 0, (size_t)NN * NH * 4, stream);
    gat_edge_max_kernel<<<cdiv((long long)ET * NH, 256), T, 0, stream>>>(ei, ssrc, sdst, emax);
    gat_edge_exp_kernel<<<cdiv((long long)ET * NH, 256), T, 0, stream>>>(ei, ssrc, sdst, emax, exbuf, denom);
    hipMemsetAsync(agg, 0, (size_t)NN * HIDN * 4, stream);
    gat_edge_agg_kernel<<<cdiv((long long)ET * NH, 256), T, 0, stream>>>(ei, hraw, exbuf, denom, agg);
    bias_elu_bf16_kernel<<<cdiv((long long)NN * HIDN, 256), T, 0, stream>>>(agg, gb[l], hact);
  }

  // ---- output projection: go = gelu(h@pw1+pb1)@pw2+pb2
  {
    dim3 G1(HIDN / 64, cdiv(NN, 128));
    gemm_wmma_kernel<2, 1><<<G1, T, 0, stream>>>(hact, pw1t, pb1, t1b, NN, HIDN, HIDN, HIDN, HIDN, HIDN);
    dim3 G2(DD / 64, cdiv(NN, 128));
    gemm_wmma_kernel<1, 1><<<G2, T, 0, stream>>>(t1b, pw2t, pb2, gob, NN, DD, HIDN, HIDN, HIDN, DD);
  }

  // ---- q, k, v projections
  {
    dim3 Gq(DD / 64, cdiv(NQQ, 128));
    gemm_wmma_kernel<1, 1><<<Gq, T, 0, stream>>>(qinb, wqt, bq, qb, NQQ, DD, DD, DD, DD, DD);
    dim3 Gk(DD / 64, cdiv(NN, 128));
    gemm_wmma_kernel<1, 1><<<Gk, T, 0, stream>>>(gob, wkt, bk, kbf, NN, DD, DD, DD, DD, DD);
    gemm_wmma_kernel<1, 2><<<Gk, T, 0, stream>>>(gob, wvt, bv, vt, NN, DD, DD, DD, DD, KPAD); // transposed out
  }

  // ---- flash cross-attention
  {
    dim3 Ga(NH, cdiv(NQQ, 16));
    flash_attn_kernel<<<Ga, dim3(320), 0, stream>>>(qb, kbf, vt, ctxb);
  }

  // ---- output proj + LN1
  {
    dim3 Go(DD / 64, cdiv(NQQ, 128));
    gemm_wmma_kernel<1, 0><<<Go, T, 0, stream>>>(ctxb, wot, bo, attnout, NQQ, DD, DD, DD, DD, DD);
    add_ln_kernel<<<NQQ, T, 0, stream>>>(queries, attnout, ln1g, ln1b, a1, a1b);
  }

  // ---- FFN + LN2 -> out
  {
    dim3 G1(4 * DD / 64, cdiv(NQQ, 128));
    gemm_wmma_kernel<2, 1><<<G1, T, 0, stream>>>(a1b, fw1t, fb1, f1b, NQQ, 4 * DD, DD, DD, DD, 4 * DD);
    dim3 G2(DD / 64, cdiv(NQQ, 128));
    gemm_wmma_kernel<1, 0><<<G2, T, 0, stream>>>(f1b, fw2t, fb2, ffn2, NQQ, DD, 4 * DD, 4 * DD, 4 * DD, DD);
    add_ln_kernel<<<NQQ, T, 0, stream>>>(a1, ffn2, ln2g, ln2b, out, nullptr);
  }
}
